// Model_47588237639843
// MI455X (gfx1250) — matrix operand order repaired and checked by reading, not compiled, not run on hardware
//
#include <hip/hip_runtime.h>

// CRF NLL for B=256, T=1024, K=128 on gfx1250.
// Forward recursion rewritten in the exp-domain so the K x K contraction runs
// on v_wmma_f32_16x16x32_f16:
//   alpha' = log( (p @ E^T) * exp(emit + wmax) ) + m,   p = exp(alpha - m) in [0,1]
// E = exp(W - rowmax(W)) in [exp(-8),1]  -> both WMMA operands f16-safe.
// Emit loads are software-pipelined one step ahead (registers) + prefetched
// two steps ahead (GL2) to keep global latency off the sequential scan path.

typedef __attribute__((ext_vector_type(16))) _Float16 v16h;
typedef __attribute__((ext_vector_type(8)))  float    v8f;

#define Tk 1024
#define Kk 128
#define EPAD 136   // f16 row stride for E and p (pad vs 64-bank LDS)
#define XPAD 132   // f32 row stride for staged q

__global__ __launch_bounds__(256) void crf_forward_kernel(
    const float* __restrict__ logits,
    const float* __restrict__ W,
    const int*  __restrict__ seq_len,
    float* __restrict__ ws_third)
{
  __shared__ __align__(16) _Float16 sEW[Kk * EPAD];  // E[j][i] = exp(W[j,i]-wmax[j])
  __shared__ __align__(16) _Float16 sP [16 * EPAD];  // p[b][i] f16
  __shared__ float sX[16 * XPAD];                    // staged q
  __shared__ float sWmax[Kk];
  __shared__ float sM[16];                           // running log-offset m[b]
  __shared__ float sLse[16];

  const int tid  = threadIdx.x;
  const int wg   = blockIdx.x;      // 0..15 -> batch tile
  const int bg0  = wg * 16;
  const int lane = tid & 31;
  const int wave = tid >> 5;        // 0..7 -> label tile (N)
  const int hi   = lane >> 4;
  const int lo   = lane & 15;

  // ---------------- setup: wmax + E (one-time) ----------------
  if (tid < Kk) {
    const float* wr = W + tid * Kk;
    float mx = wr[0];
    for (int i = 1; i < Kk; ++i) mx = fmaxf(mx, wr[i]);
    sWmax[tid] = mx;
    for (int i = 0; i < Kk; ++i)
      sEW[tid * EPAD + i] = (_Float16)__expf(wr[i] - mx);
  }
  __syncthreads();

  // ---- resident B fragments: B[k][n] = E[n][k]  (ISA 32x16 f16 B layout:
  //      lanes 0-15 hold K{0..7,16..23}, lanes 16-31 hold K{8..15,24..31})
  v16h bfrag[4];
  {
    const int N = wave * 16 + lo;
    for (int f = 0; f < 4; ++f) {
      union { uint4 u[2]; v16h h; } cv;
      const _Float16* src = sEW + N * EPAD + 32 * f + 8 * hi;
      cv.u[0] = *(const uint4*)(src);
      cv.u[1] = *(const uint4*)(src + 16);
      bfrag[f] = cv.h;
    }
  }

  // ---------------- reduction-phase thread roles ----------------
  const int row = tid >> 4;           // batch row in tile
  const int cs  = tid & 15;           // 8-column segment
  const int slen = seq_len[bg0 + row];
  const long rowbase0 = ((long)(bg0 + row) * Tk) * Kk;  // logits[b][0][:]

  // ---------------- init: alpha0 = logits[:,0,:] ----------------
  {
    float xv[8], lm = -3.4e38f;
    for (int k2 = 0; k2 < 8; ++k2) {
      xv[k2] = logits[rowbase0 + cs * 8 + k2];
      lm = fmaxf(lm, xv[k2]);
    }
    for (int d = 1; d < 16; d <<= 1) lm = fmaxf(lm, __shfl_xor(lm, d, 32));
    for (int k2 = 0; k2 < 8; ++k2)
      sP[row * EPAD + cs * 8 + k2] = (_Float16)__expf(xv[k2] - lm);
    if (cs == 0) sM[row] = lm;
  }
  __syncthreads();

  // ---------------- per-lane compute-phase constants ----------------
  const int jlane = wave * 16 + lo;        // output label index this lane owns
  const float wmj = sWmax[jlane];
  long lbase[8];                           // logits base per accumulator row
  for (int r = 0; r < 8; ++r)
    lbase[r] = ((long)(bg0 + r + 8 * hi) * Tk) * Kk + jlane;

  // software-pipelined emit registers: preload step t=1
  float ecur[8];
  for (int r = 0; r < 8; ++r)
    ecur[r] = logits[lbase[r] + (long)Kk];

  // ---------------- time scan ----------------
  for (int t = 1; t < Tk; ++t) {
    // issue next step's emit loads + prefetch two steps ahead (GL2)
    float enext[8];
    if (t + 1 < Tk) {
      for (int r = 0; r < 8; ++r) {
        enext[r] = logits[lbase[r] + (long)(t + 1) * Kk];
        __builtin_prefetch(&logits[lbase[r] + (long)(t + 2) * Kk], 0, 0);
      }
    }

    // A fragments from sP (ISA 16x32 f16 A layout: lane=M, halves =
    // K{0..7,16..23} for lanes 0-15 / K{8..15,24..31} for lanes 16-31)
    v8f acc = {};
    for (int f = 0; f < 4; ++f) {
      union { uint4 u[2]; v16h h; } cv;
      const _Float16* src = sP + lo * EPAD + 32 * f + 8 * hi;
      cv.u[0] = *(const uint4*)(src);
      cv.u[1] = *(const uint4*)(src + 16);
      acc = __builtin_amdgcn_wmma_f32_16x16x32_f16(false, cv.h, false, bfrag[f],
                                                   (short)0, acc, false, false);
    }
    // q[b][j] = (p @ E^T) * exp(emit + wmax[j]); stage for row-renormalize
    for (int r = 0; r < 8; ++r)
      sX[(r + 8 * hi) * XPAD + jlane] = acc[r] * __expf(ecur[r] + wmj);
    __syncthreads();

    // masked renormalize: p' = q / max(q), m += log(max(q))
    {
      float xv[8], lm = -3.4e38f;
      for (int k2 = 0; k2 < 8; ++k2) {
        xv[k2] = sX[row * XPAD + cs * 8 + k2];
        lm = fmaxf(lm, xv[k2]);
      }
      for (int d = 1; d < 16; d <<= 1) lm = fmaxf(lm, __shfl_xor(lm, d, 32));
      if (t < slen) {
        float inv = 1.0f / lm;
        for (int k2 = 0; k2 < 8; ++k2)
          sP[row * EPAD + cs * 8 + k2] = (_Float16)(xv[k2] * inv);
        if (cs == 0) sM[row] += __logf(lm);
      }
    }
    __syncthreads();

    for (int r = 0; r < 8; ++r) ecur[r] = enext[r];
  }

  // ---------------- third term: sum_b logsumexp(alpha) = m + log(sum p) ----
  {
    float s = 0.f;
    for (int k2 = 0; k2 < 8; ++k2)
      s += (float)sP[row * EPAD + cs * 8 + k2];
    for (int d = 1; d < 16; d <<= 1) s += __shfl_xor(s, d, 32);
    if (cs == 0) sLse[row] = sM[row] + __logf(s);
  }
  __syncthreads();
  if (tid == 0) {
    float th = 0.f;
    for (int r = 0; r < 16; ++r) th += sLse[r];
    ws_third[wg] = th;   // deterministic fixed-slot partial
  }
}

// unary (first) + transition (second) terms, one block per batch
__global__ __launch_bounds__(256) void crf_pointwise_kernel(
    const float* __restrict__ logits, const float* __restrict__ W,
    const int* __restrict__ gold, const int* __restrict__ seq_len,
    float* __restrict__ ws_fs)
{
  __shared__ float red[256];
  const int b = blockIdx.x, tid = threadIdx.x;
  const int sl = seq_len[b];
  float s = 0.f;
  for (int t = tid; t < Tk; t += 256) {
    int g = gold[b * Tk + t];
    if (t < sl)     s += logits[((long)b * Tk + t) * Kk + g];
    if (t < sl - 1) s += W[g * Kk + gold[b * Tk + t + 1]];
  }
  red[tid] = s;
  __syncthreads();
  for (int o = 128; o > 0; o >>= 1) {   // fixed-order tree: deterministic
    if (tid < o) red[tid] += red[tid + o];
    __syncthreads();
  }
  if (tid == 0) ws_fs[b] = red[0];
}

__global__ void crf_finalize_kernel(const float* __restrict__ ws,
                                    float* __restrict__ out)
{
  if (threadIdx.x == 0 && blockIdx.x == 0) {
    float th = 0.f, fs = 0.f;
    for (int i = 0; i < 16; ++i)  th += ws[i];
    for (int i = 0; i < 256; ++i) fs += ws[16 + i];
    out[0] = -(fs - th) / 256.0f;
  }
}

extern "C" void kernel_launch(void* const* d_in, const int* in_sizes, int n_in,
                              void* d_out, int out_size, void* d_ws, size_t ws_size,
                              hipStream_t stream) {
  const float* logits  = (const float*)d_in[0];
  const float* W       = (const float*)d_in[1];
  const int*   gold    = (const int*)d_in[2];
  const int*   seq_len = (const int*)d_in[3];
  float* ws = (float*)d_ws;   // [0..15]=third partials, [16..271]=fs partials

  crf_forward_kernel  <<<16,  256, 0, stream>>>(logits, W, seq_len, ws);
  crf_pointwise_kernel<<<256, 256, 0, stream>>>(logits, W, gold, seq_len, ws + 16);
  crf_finalize_kernel <<<1,   32,  0, stream>>>(ws, (float*)d_out);
}
